// Model_39676907881791
// MI455X (gfx1250) — compile-verified, matrix operand order checked by reading, not run on hardware
//
#include <hip/hip_runtime.h>
#include <hip/hip_bf16.h>

// Flash-attention forward: B=4,H=16,L=2048,D=128, fp32 in/out, scale=1.0.
// QK^T and PV on v_wmma_f32_16x16x32_f16 (fp32 accum). Row sums via an extra
// WMMA against an all-ones B tile (l stays lane-local, no shuffle tree).
// 512-thread workgroups (16 wave32): K/V tile staged once per 256 query rows,
// while each wave keeps the <256-VGPR footprint (no s_set_vgpr_msb churn).

typedef __attribute__((ext_vector_type(16))) _Float16 v16h;
typedef __attribute__((ext_vector_type(8)))  _Float16 v8h;
typedef __attribute__((ext_vector_type(2)))  _Float16 v2h;
typedef __attribute__((ext_vector_type(8)))  float    v8f;

namespace {
constexpr int L  = 2048;
constexpr int D  = 128;
constexpr int BH = 64;    // B*H
constexpr int BQ = 256;   // query rows per workgroup (16 waves x 16 rows)
constexpr int BK = 64;    // keys per main-loop iteration
constexpr int WAVES = 16;
constexpr int NTHREADS = WAVES * 32;   // 512
}

__global__ __launch_bounds__(NTHREADS, 1)
void fa_fwd_f32(const float* __restrict__ Q,
                const float* __restrict__ K,
                const float* __restrict__ V,
                float* __restrict__ O)
{
    __shared__ __align__(16) _Float16 sK [BK * D];         // [key][d]  row-major
    __shared__ __align__(16) _Float16 sVt[D * BK];         // [d][key]  transposed
    __shared__ __align__(16) _Float16 sP [WAVES][16 * BK]; // per-wave P staging

    const int tid  = threadIdx.x;
    const int wave = tid >> 5;
    const int lane = tid & 31;
    const int bh   = blockIdx.y;
    const int qrow0 = blockIdx.x * BQ + wave * 16;

    const float* qb = Q + (size_t)bh * L * D + (size_t)qrow0 * D;
    const float* kb = K + (size_t)bh * L * D;
    const float* vb = V + (size_t)bh * L * D;
    float*       ob = O + (size_t)bh * L * D + (size_t)qrow0 * D;

    const int m     = lane & 15;            // A-layout row / B-layout column
    const int off   = (lane < 16) ? 0 : 8;  // half-wave K-element offset (ISA layout)
    const int rbase = (lane >> 4) * 8;      // C-layout row base for this lane

    // ---- Q tile (16 x 128) -> registers in exact 16-bit A-matrix layout ----
    v16h a_q[4];
#pragma unroll
    for (int c = 0; c < 4; ++c) {
        const float* p0 = qb + m * D + c * 32 + off;       // K = c*32 + off + e
        const float* p1 = p0 + 16;                         // K = c*32 + 16 + off + e
        v16h a;
#pragma unroll
        for (int e = 0; e < 8; ++e) {
            a[e]     = (_Float16)p0[e];
            a[e + 8] = (_Float16)p1[e];
        }
        a_q[c] = a;
    }

    // all-ones B tile: rowsum(P) via WMMA (every output column == row sum)
    v16h b_ones;
#pragma unroll
    for (int e = 0; e < 16; ++e) b_ones[e] = (_Float16)1.0f;

    v8f acc[8];                 // O accumulator, 8 d-chunks
#pragma unroll
    for (int n = 0; n < 8; ++n) acc[n] = (v8f){};
    v8f acc_l = (v8f){};        // row-sum accumulator (ones-column tile)
    float mrow[8];
#pragma unroll
    for (int r = 0; r < 8; ++r) mrow[r] = -1e30f;

    v2h* const sK2  = (v2h*)sK;     // [key][d-pair]
    v2h* const sVt2 = (v2h*)sVt;    // [d][key-pair]

    for (int kb0 = 0; kb0 < L; kb0 += BK) {
        __syncthreads();
        // ---- stage K: fp32 d-pairs -> packed f16 b32 LDS stores ----
#pragma unroll
        for (int i = 0; i < 8; ++i) {
            int idx = i * NTHREADS + tid;       // 0..4095 (64 keys x 64 d-pairs)
            int key = idx >> 6;
            int dp  = idx & 63;
            float2 kv = *(const float2*)(kb + (size_t)(kb0 + key) * D + 2 * dp);
            v2h p; p[0] = (_Float16)kv.x; p[1] = (_Float16)kv.y;
            sK2[key * 64 + dp] = p;
        }
        // ---- stage V transposed: two consecutive keys packed per b32 store ----
#pragma unroll
        for (int i = 0; i < 8; ++i) {
            int idx = i * NTHREADS + tid;       // 0..4095 (32 key-pairs x 128 d)
            int kp  = idx >> 7;
            int d   = idx & 127;
            float a = vb[(size_t)(kb0 + 2 * kp) * D + d];
            float b = vb[(size_t)(kb0 + 2 * kp + 1) * D + d];
            v2h p; p[0] = (_Float16)a; p[1] = (_Float16)b;
            sVt2[d * 32 + kp] = p;              // sVt[d*64 + 2kp]
        }
        if (kb0 + BK < L) {                     // global_prefetch_b8 next tiles
            __builtin_prefetch(&kb[(size_t)(kb0 + BK) * D + tid * 16], 0, 0);
            __builtin_prefetch(&vb[(size_t)(kb0 + BK) * D + tid * 16], 0, 0);
        }
        __syncthreads();

        // ---- S = Q * K^T : four 16-key column blocks, 4 d-chunks each ----
        v8f S[4];
#pragma unroll
        for (int cb = 0; cb < 4; ++cb) {
            v8f s = (v8f){};
            const int key = cb * 16 + m;        // B-matrix: lane -> N (key)
#pragma unroll
            for (int c = 0; c < 4; ++c) {
                const _Float16* bp = &sK[key * D + c * 32 + off];
                v8h lo = *(const v8h*)bp;        // K = c*32 + off + e
                v8h hi = *(const v8h*)(bp + 16); // K = c*32 + 16 + off + e
                v16h b;
#pragma unroll
                for (int e = 0; e < 8; ++e) { b[e] = lo[e]; b[e + 8] = hi[e]; }
                s = __builtin_amdgcn_wmma_f32_16x16x32_f16(
                        false, a_q[c], false, b, (short)0, s, false, false);
            }
            S[cb] = s;
        }

        // ---- online softmax: row max (shuffle tree), P = exp(S - m) ----
        float alpha[8];
#pragma unroll
        for (int r = 0; r < 8; ++r) {
            float x = fmaxf(fmaxf(S[0][r], S[1][r]), fmaxf(S[2][r], S[3][r]));
#pragma unroll
            for (int xm = 8; xm >= 1; xm >>= 1)
                x = fmaxf(x, __shfl_xor(x, xm, 32));   // within 16-lane half
            float mn = fmaxf(mrow[r], x);
            alpha[r] = __expf(mrow[r] - mn);
            mrow[r]  = mn;
#pragma unroll
            for (int cb = 0; cb < 4; ++cb)
                S[cb][r] = __expf(S[cb][r] - mn);
        }
        // rescale all accumulators (O chunks + row-sum tile)
#pragma unroll
        for (int r = 0; r < 8; ++r) {
            acc_l[r] *= alpha[r];
#pragma unroll
            for (int n = 0; n < 8; ++n) acc[n][r] *= alpha[r];
        }

        // ---- re-layout P: C-layout -> row-major LDS -> A-matrix layout ----
#pragma unroll
        for (int cb = 0; cb < 4; ++cb)
#pragma unroll
            for (int r = 0; r < 8; ++r)
                sP[wave][(rbase + r) * BK + cb * 16 + m] = (_Float16)S[cb][r];

        v16h a_p[2];                            // K = 64 -> two A chunks
#pragma unroll
        for (int h = 0; h < 2; ++h) {
            const _Float16* pp = &sP[wave][m * BK + h * 32 + off];
            v8h lo = *(const v8h*)pp;
            v8h hi = *(const v8h*)(pp + 16);
            v16h a;
#pragma unroll
            for (int e = 0; e < 8; ++e) { a[e] = lo[e]; a[e + 8] = hi[e]; }
            a_p[h] = a;
        }

        // ---- O += P * V ; l += P * ones ----
#pragma unroll
        for (int h = 0; h < 2; ++h) {
#pragma unroll
            for (int n = 0; n < 8; ++n) {
                const int dcol = n * 16 + m;    // B-matrix: lane -> N (d column)
                const _Float16* bp = &sVt[dcol * BK + h * 32 + off];
                v8h lo = *(const v8h*)bp;       // keys h*32 + off + e
                v8h hi = *(const v8h*)(bp + 16);
                v16h b;
#pragma unroll
                for (int e = 0; e < 8; ++e) { b[e] = lo[e]; b[e + 8] = hi[e]; }
                acc[n] = __builtin_amdgcn_wmma_f32_16x16x32_f16(
                             false, a_p[h], false, b, (short)0, acc[n], false, false);
            }
            acc_l = __builtin_amdgcn_wmma_f32_16x16x32_f16(
                        false, a_p[h], false, b_ones, (short)0, acc_l, false, false);
        }
    }

    // ---- normalize and store fp32 output (l is lane-local in acc_l) ----
    float inv[8];
#pragma unroll
    for (int r = 0; r < 8; ++r) inv[r] = 1.0f / acc_l[r];
#pragma unroll
    for (int n = 0; n < 8; ++n)
#pragma unroll
        for (int r = 0; r < 8; ++r)
            ob[(size_t)(rbase + r) * D + n * 16 + m] = acc[n][r] * inv[r];
}

extern "C" void kernel_launch(void* const* d_in, const int* in_sizes, int n_in,
                              void* d_out, int out_size, void* d_ws, size_t ws_size,
                              hipStream_t stream)
{
    (void)in_sizes; (void)n_in; (void)out_size; (void)d_ws; (void)ws_size;
    const float* q = (const float*)d_in[0];
    const float* k = (const float*)d_in[1];
    const float* v = (const float*)d_in[2];
    float* out = (float*)d_out;

    dim3 grid(L / BQ, BH);     // 8 x 64 workgroups
    dim3 block(NTHREADS);      // 16 wave32
    fa_fwd_f32<<<grid, block, 0, stream>>>(q, k, v, out);
}
